// Classifier_52965536694535
// MI455X (gfx1250) — matrix operand order repaired and checked by reading, not compiled, not run on hardware
//
#include <hip/hip_runtime.h>

typedef __bf16 bf16;
typedef __attribute__((ext_vector_type(4)))  __bf16 v4bf;
typedef __attribute__((ext_vector_type(8)))  __bf16 v8bf;
typedef __attribute__((ext_vector_type(16))) __bf16 v16bf;
typedef __attribute__((ext_vector_type(8)))  float  v8f;

// Problem dims (match reference)
constexpr int BATCH = 16384;
constexpr int FDIM  = 2048;
constexpr int HDIM  = 4096;
constexpr int DDEM  = 4096;
constexpr int KFEAT = 2048;   // power of two (used for &)

// ---------------------------------------------------------------------------
// CDNA5 async global->LDS copy: 16B per lane, no VGPR data round-trip.
// Tracked by ASYNCcnt (ISA 15.18.3 op 98 GLOBAL_LOAD_ASYNC_TO_LDS_B128).
// ---------------------------------------------------------------------------
__device__ __forceinline__ void async_copy_b128(unsigned lds_addr,
                                                const void* gaddr) {
  asm volatile("global_load_async_to_lds_b128 %0, %1, off"
               :: "v"(lds_addr), "v"(gaddr)
               : "memory");
}
__device__ __forceinline__ void wait_asynccnt0() {
  asm volatile("s_wait_asynccnt 0x0" ::: "memory");
}

// ---------------------------------------------------------------------------
// Hinge loss: partial sums (deterministic two-stage reduction, no atomics)
// ---------------------------------------------------------------------------
__global__ __launch_bounds__(256) void subdom_partial_kernel(
    const float* __restrict__ demo_metric,
    const float* __restrict__ alpha,
    const float* __restrict__ sample_loss,
    const float* __restrict__ subc,
    float* __restrict__ partial,
    long long total)
{
  const float c = subc[0];
  float acc = 0.f;
  const long long n4 = total >> 2;
  const long long stride = (long long)gridDim.x * blockDim.x;
  for (long long i = (long long)blockIdx.x * blockDim.x + threadIdx.x;
       i < n4; i += stride) {
    const float4 sl = ((const float4*)sample_loss)[i];
    const float4 dm = ((const float4*)demo_metric)[i];
    const int k = (int)((i << 2) & (long long)(KFEAT - 1));
    const float4 al = *(const float4*)(alpha + k);
    acc += fmaxf(fmaf(al.x, sl.x - dm.x, 1.f), 0.f) - c;
    acc += fmaxf(fmaf(al.y, sl.y - dm.y, 1.f), 0.f) - c;
    acc += fmaxf(fmaf(al.z, sl.z - dm.z, 1.f), 0.f) - c;
    acc += fmaxf(fmaf(al.w, sl.w - dm.w, 1.f), 0.f) - c;
  }
  __shared__ float red[256];
  red[threadIdx.x] = acc;
  __syncthreads();
  #pragma unroll
  for (int s = 128; s > 0; s >>= 1) {
    if ((int)threadIdx.x < s) red[threadIdx.x] += red[threadIdx.x + s];
    __syncthreads();
  }
  if (threadIdx.x == 0) partial[blockIdx.x] = red[0];
}

__global__ __launch_bounds__(256) void reduce_partial_kernel(
    const float* __restrict__ partial, int n, float* __restrict__ out)
{
  float acc = 0.f;
  for (int i = threadIdx.x; i < n; i += 256) acc += partial[i];
  __shared__ float red[256];
  red[threadIdx.x] = acc;
  __syncthreads();
  #pragma unroll
  for (int s = 128; s > 0; s >>= 1) {
    if ((int)threadIdx.x < s) red[threadIdx.x] += red[threadIdx.x + s];
    __syncthreads();
  }
  if (threadIdx.x == 0) out[0] = red[0];
}

// ---------------------------------------------------------------------------
// f32 -> bf16 elementwise convert (x), 4-wide
// ---------------------------------------------------------------------------
__global__ __launch_bounds__(256) void cvt_f32_bf16_kernel(
    const float* __restrict__ in, bf16* __restrict__ out, long long n4)
{
  const long long i = (long long)blockIdx.x * blockDim.x + threadIdx.x;
  if (i >= n4) return;
  const float4 v = ((const float4*)in)[i];
  v4bf o;
  o[0] = (bf16)v.x; o[1] = (bf16)v.y; o[2] = (bf16)v.z; o[3] = (bf16)v.w;
  ((v4bf*)out)[i] = o;
}

// ---------------------------------------------------------------------------
// W [K x N] f32 -> Wt [N x K] bf16 (tiled transpose via LDS)
// ---------------------------------------------------------------------------
__global__ __launch_bounds__(256) void transpose_f32_bf16_kernel(
    const float* __restrict__ W, bf16* __restrict__ Wt, int K, int N)
{
  __shared__ float tile[32][33];
  const int nb = blockIdx.x * 32;           // column base (N)
  const int kb = blockIdx.y * 32;           // row base (K)
  const int tx = threadIdx.x & 31;
  const int ty = threadIdx.x >> 5;          // 0..7
  #pragma unroll
  for (int i = 0; i < 32; i += 8)
    tile[ty + i][tx] = W[(size_t)(kb + ty + i) * N + (nb + tx)];
  __syncthreads();
  #pragma unroll
  for (int i = 0; i < 32; i += 8)
    Wt[(size_t)(nb + ty + i) * K + (kb + tx)] = (bf16)tile[tx][ty + i];
}

// ---------------------------------------------------------------------------
// GEMM: C[M,N] = relu(A[M,K] @ Bt[N,K]^T + bias), bf16 in, f32 acc, bf16 out
// Block: 256 thr (8 waves), tile 64(M) x 256(N), BK=32.
// Wave grid 2(M) x 4(N); each wave: 2x4 tiles -> 8 x v_wmma_f32_16x16x32_bf16.
// Double-buffered LDS fed by GLOBAL_LOAD_ASYNC_TO_LDS_B128 (ASYNCcnt):
// no staging VGPRs -> no scratch spills; one barrier per K-step.
// LDS row stride 48 bf16 (96B): 16B-aligned b128 ops; total 60KB (2 buffers).
// ---------------------------------------------------------------------------
constexpr int BM = 64, BN = 256, BK = 32;
constexpr int LDSS = 48;
constexpr int ABUF = BM * LDSS;   // elems per A buffer
constexpr int BBUF = BN * LDSS;   // elems per B buffer

__global__ __launch_bounds__(256, 2) void gemm_bf16_relu_kernel(
    const bf16* __restrict__ A,
    const bf16* __restrict__ Bt,
    const float* __restrict__ bias,
    bf16* __restrict__ C,
    int M, int N, int K)
{
  __shared__ bf16 sA[2 * ABUF];
  __shared__ bf16 sB[2 * BBUF];

  const int tid  = threadIdx.x;
  const int lane = tid & 31;
  const int wave = tid >> 5;
  const int wm   = wave & 1;          // 0..1  -> 32 rows each
  const int wn   = wave >> 1;         // 0..3  -> 64 cols each

  const int block_m = blockIdx.y * BM;
  const int block_n = blockIdx.x * BN;

  v8f acc[2][4] = {};

  const int frow  = lane & 15;
  const int ahalf = lane >> 4;                 // 0/1
  const int akoff = ahalf << 3;                // A frag: K {0..7}/{8..15} (+16)
  const int bkoff = ahalf << 3;                // B frag: K {0..7}/{8..15} (+16)

  // staging indices: thread -> (row, 16B chunk)
  const int ar = tid >> 2;                     // 0..63
  const int ac = (tid & 3) << 3;               // 0,8,16,24

  const bf16* aptr = A + (size_t)(block_m + ar) * K + ac;
  const bf16* bptr = Bt + (size_t)(block_n + ar) * K + ac;
  const size_t brow_stride = (size_t)64 * K;   // 64 Bt rows per chunk group

  // per-thread LDS destinations (byte addresses), per buffer
  unsigned ldsA[2], ldsB[2];
  #pragma unroll
  for (int b = 0; b < 2; ++b) {
    ldsA[b] = (unsigned)(size_t)(&sA[b * ABUF + ar * LDSS + ac]);
    ldsB[b] = (unsigned)(size_t)(&sB[b * BBUF + ar * LDSS + ac]);
  }
  constexpr unsigned BCH = 64u * LDSS * 2u;    // bytes between B chunk groups

  const int nsteps = K / BK;

  // stage tile 0 into buffer 0
  async_copy_b128(ldsA[0], aptr);
  #pragma unroll
  for (int i = 0; i < 4; ++i)
    async_copy_b128(ldsB[0] + (unsigned)i * BCH, bptr + (size_t)i * brow_stride);
  wait_asynccnt0();
  __syncthreads();

  for (int s = 0; s < nsteps; ++s) {
    const int buf = s & 1;
    const bool more = (s + 1) < nsteps;

    // kick off next tile's async DMA into the other buffer (hidden by WMMA)
    if (more) {
      const int k0n = (s + 1) * BK;
      const int nbuf = buf ^ 1;
      async_copy_b128(ldsA[nbuf], aptr + k0n);
      #pragma unroll
      for (int i = 0; i < 4; ++i)
        async_copy_b128(ldsB[nbuf] + (unsigned)i * BCH,
                        bptr + (size_t)i * brow_stride + k0n);
    }

    // fragments from current buffer
    v16bf af[2], bfr[4];
    #pragma unroll
    for (int i = 0; i < 2; ++i) {
      const bf16* p = &sA[buf * ABUF + (wm * 32 + i * 16 + frow) * LDSS];
      v8bf lo = *(const v8bf*)(p + akoff);
      v8bf hi = *(const v8bf*)(p + 16 + akoff);
      af[i] = __builtin_shufflevector(lo, hi,
              0,1,2,3,4,5,6,7,8,9,10,11,12,13,14,15);
    }
    #pragma unroll
    for (int j = 0; j < 4; ++j) {
      const bf16* p = &sB[buf * BBUF + (wn * 64 + j * 16 + frow) * LDSS];
      v8bf lo = *(const v8bf*)(p + bkoff);
      v8bf hi = *(const v8bf*)(p + 16 + bkoff);
      bfr[j] = __builtin_shufflevector(lo, hi,
               0,1,2,3,4,5,6,7,8,9,10,11,12,13,14,15);
    }
    #pragma unroll
    for (int i = 0; i < 2; ++i)
      #pragma unroll
      for (int j = 0; j < 4; ++j)
        acc[i][j] = __builtin_amdgcn_wmma_f32_16x16x32_bf16(
            false, af[i], false, bfr[j], (short)0, acc[i][j], false, false);

    // next buffer fully landed + all waves done reading current buffer
    wait_asynccnt0();
    __syncthreads();
  }

  // epilogue: bias + relu, store bf16
  #pragma unroll
  for (int j = 0; j < 4; ++j) {
    const int n  = block_n + wn * 64 + j * 16 + frow;
    const float bv = bias[n];
    #pragma unroll
    for (int i = 0; i < 2; ++i) {
      const int mbase = block_m + wm * 32 + i * 16 + (ahalf << 3);
      #pragma unroll
      for (int v = 0; v < 8; ++v) {
        float r = acc[i][j][v] + bv;
        r = fmaxf(r, 0.f);
        C[(size_t)(mbase + v) * N + n] = (bf16)r;
      }
    }
  }
}

// ---------------------------------------------------------------------------
// Final layer: probs = sigmoid(h3 @ W4 + b4), one wave per row, N=2
// ---------------------------------------------------------------------------
__global__ __launch_bounds__(256) void head_kernel(
    const bf16* __restrict__ h3, const float* __restrict__ W4,
    const float* __restrict__ b4, float* __restrict__ probs, int Brows, int H)
{
  const int wave = threadIdx.x >> 5;
  const int lane = threadIdx.x & 31;
  const int row  = blockIdx.x * 8 + wave;
  if (row >= Brows) return;
  const bf16* hp = h3 + (size_t)row * H;
  float a0 = 0.f, a1 = 0.f;
  for (int base = lane * 8; base < H; base += 32 * 8) {
    const v8bf hv = *(const v8bf*)(hp + base);
    #pragma unroll
    for (int u = 0; u < 8; ++u) {
      const float h = (float)hv[u];
      a0 = fmaf(h, W4[2 * (base + u) + 0], a0);
      a1 = fmaf(h, W4[2 * (base + u) + 1], a1);
    }
  }
  #pragma unroll
  for (int off = 16; off > 0; off >>= 1) {
    a0 += __shfl_down(a0, off);
    a1 += __shfl_down(a1, off);
  }
  if (lane == 0) {
    const float l0 = a0 + b4[0];
    const float l1 = a1 + b4[1];
    probs[2 * row + 0] = 1.f / (1.f + __expf(-l0));
    probs[2 * row + 1] = 1.f / (1.f + __expf(-l1));
  }
}

// ---------------------------------------------------------------------------
// Workspace layout (bytes)
// ---------------------------------------------------------------------------
constexpr size_t SZ_XB  = (size_t)BATCH * FDIM * 2;   //  64 MB  x bf16
constexpr size_t SZ_W1T = (size_t)HDIM * FDIM * 2;    //  16 MB
constexpr size_t SZ_WHT = (size_t)HDIM * HDIM * 2;    //  32 MB
constexpr size_t SZ_H   = (size_t)BATCH * HDIM * 2;   // 128 MB
constexpr size_t OFF_XB   = 0;
constexpr size_t OFF_W1T  = OFF_XB  + SZ_XB;
constexpr size_t OFF_W2T  = OFF_W1T + SZ_W1T;
constexpr size_t OFF_W3T  = OFF_W2T + SZ_WHT;
constexpr size_t OFF_H1   = OFF_W3T + SZ_WHT;
constexpr size_t OFF_H2   = OFF_H1  + SZ_H;
constexpr size_t OFF_PART = OFF_H2  + SZ_H;           // 1024 floats
constexpr int    NPART    = 1024;

extern "C" void kernel_launch(void* const* d_in, const int* in_sizes, int n_in,
                              void* d_out, int out_size, void* d_ws, size_t ws_size,
                              hipStream_t stream)
{
  (void)in_sizes; (void)n_in; (void)out_size; (void)ws_size;

  const float* x           = (const float*)d_in[0];
  const float* demo_metric = (const float*)d_in[1];
  const float* alpha       = (const float*)d_in[2];
  const float* sample_loss = (const float*)d_in[3];
  const float* subc        = (const float*)d_in[4];
  // d_in[5], d_in[6]: num_of_demos / num_of_features (compile-time here)
  const float* W1 = (const float*)d_in[7];
  const float* b1 = (const float*)d_in[8];
  const float* W2 = (const float*)d_in[9];
  const float* b2 = (const float*)d_in[10];
  const float* W3 = (const float*)d_in[11];
  const float* b3 = (const float*)d_in[12];
  const float* W4 = (const float*)d_in[13];
  const float* b4 = (const float*)d_in[14];

  float* out = (float*)d_out;          // out[0] = loss, out[1..] = probs

  char* ws = (char*)d_ws;
  bf16*  xb   = (bf16*)(ws + OFF_XB);
  bf16*  w1t  = (bf16*)(ws + OFF_W1T);
  bf16*  w2t  = (bf16*)(ws + OFF_W2T);
  bf16*  w3t  = (bf16*)(ws + OFF_W3T);
  bf16*  h1   = (bf16*)(ws + OFF_H1);
  bf16*  h2   = (bf16*)(ws + OFF_H2);
  bf16*  h3   = h1;                    // reuse: h1 dead once h2 is built
  float* part = (float*)(ws + OFF_PART);

  // ---- hinge loss (independent of MLP) ----
  const long long total = (long long)DDEM * KFEAT;
  subdom_partial_kernel<<<NPART, 256, 0, stream>>>(
      demo_metric, alpha, sample_loss, subc, part, total);
  reduce_partial_kernel<<<1, 256, 0, stream>>>(part, NPART, out);

  // ---- precision conversion ----
  const long long n4 = (long long)BATCH * FDIM / 4;
  cvt_f32_bf16_kernel<<<(unsigned)((n4 + 255) / 256), 256, 0, stream>>>(x, xb, n4);
  transpose_f32_bf16_kernel<<<dim3(HDIM / 32, FDIM / 32), 256, 0, stream>>>(W1, w1t, FDIM, HDIM);
  transpose_f32_bf16_kernel<<<dim3(HDIM / 32, HDIM / 32), 256, 0, stream>>>(W2, w2t, HDIM, HDIM);
  transpose_f32_bf16_kernel<<<dim3(HDIM / 32, HDIM / 32), 256, 0, stream>>>(W3, w3t, HDIM, HDIM);

  // ---- MLP (WMMA bf16, f32 accumulate) ----
  const dim3 ggrid(HDIM / BN, BATCH / BM);
  gemm_bf16_relu_kernel<<<ggrid, 256, 0, stream>>>(xb, w1t, b1, h1, BATCH, HDIM, FDIM);
  gemm_bf16_relu_kernel<<<ggrid, 256, 0, stream>>>(h1, w2t, b2, h2, BATCH, HDIM, HDIM);
  gemm_bf16_relu_kernel<<<ggrid, 256, 0, stream>>>(h2, w3t, b3, h3, BATCH, HDIM, HDIM);

  // ---- head + sigmoid ----
  head_kernel<<<BATCH / 8, 256, 0, stream>>>(h3, W4, b4, out + 1, BATCH, HDIM);
}
